// PoseModel_27367531610980
// MI455X (gfx1250) — matrix operand order repaired and checked by reading, not compiled, not run on hardware
//
#include <hip/hip_runtime.h>
#include <hip/hip_bf16.h>

#define HH 256
#define WW 256
#define NPTS 512

typedef __attribute__((ext_vector_type(16))) _Float16 v16h;
typedef __attribute__((ext_vector_type(8)))  float    v8f;

// ---------------------------------------------------------------------------
// Kernel 1: project points, bitonic depth-sort in LDS, emit
//   params[512][8]  f32 : ux, uy, conic_a, conic_b, conic_c, alpha, depth, pad
//   attrsT[16][512] f16 : channel-major WMMA B-matrix rows
//                         (ch0-2 = rgb, ch3 = depth, ch4 = 1.0, ch5-15 = 0)
// ---------------------------------------------------------------------------
__global__ __launch_bounds__(NPTS)
void prep_sort_kernel(const float* __restrict__ pc,
                      const float* __restrict__ q,
                      const float* __restrict__ t,
                      const float* __restrict__ Km,
                      const float* __restrict__ color,
                      const float* __restrict__ alpha_pt,
                      const float* __restrict__ conic,
                      float* __restrict__ params,
                      _Float16* __restrict__ attrsT)
{
    __shared__ float s_key[NPTS];
    __shared__ int   s_idx[NPTS];
    __shared__ float s_u[NPTS];
    __shared__ float s_v[NPTS];

    const int tid = threadIdx.x;

    // quaternion -> rotation (cheap, computed redundantly per thread)
    const float qx = q[0], qy = q[1], qz = q[2], qw = q[3];
    const float r00 = 1.f - 2.f*(qy*qy + qz*qz), r01 = 2.f*(qx*qy - qz*qw), r02 = 2.f*(qx*qz + qy*qw);
    const float r10 = 2.f*(qx*qy + qz*qw), r11 = 1.f - 2.f*(qx*qx + qz*qz), r12 = 2.f*(qy*qz - qx*qw);
    const float r20 = 2.f*(qx*qz - qy*qw), r21 = 2.f*(qy*qz + qx*qw), r22 = 1.f - 2.f*(qx*qx + qy*qy);

    const float px = pc[tid*3+0], py = pc[tid*3+1], pz = pc[tid*3+2];
    const float camx = t[0] + r00*px + r01*py + r02*pz;
    const float camy = t[1] + r10*px + r11*py + r12*pz;
    const float camz = t[2] + r20*px + r21*py + r22*pz;

    const float fx = Km[0], fy = Km[4], cx = Km[2], cy = Km[5];
    const float invz = 1.0f / camz;

    s_key[tid] = camz;
    s_idx[tid] = tid;
    s_u[tid]   = camx * fx * invz + cx;
    s_v[tid]   = camy * fy * invz + cy;
    __syncthreads();

    // bitonic sort (ascending by depth), key+index pairs in LDS
    for (int k = 2; k <= NPTS; k <<= 1) {
        for (int j = k >> 1; j > 0; j >>= 1) {
            const int i   = tid;
            const int ixj = i ^ j;
            if (ixj > i) {
                const bool up = ((i & k) == 0);
                const float ka = s_key[i], kb = s_key[ixj];
                if ((ka > kb) == up) {
                    s_key[i] = kb; s_key[ixj] = ka;
                    const int ia = s_idx[i]; s_idx[i] = s_idx[ixj]; s_idx[ixj] = ia;
                }
            }
            __syncthreads();
        }
    }

    const int   oi = s_idx[tid];
    const float d  = s_key[tid];

    params[tid*8+0] = s_u[oi];
    params[tid*8+1] = s_v[oi];
    params[tid*8+2] = conic[oi*3+0];
    params[tid*8+3] = conic[oi*3+1];
    params[tid*8+4] = conic[oi*3+2];
    params[tid*8+5] = alpha_pt[oi];
    params[tid*8+6] = d;
    params[tid*8+7] = 0.0f;

    attrsT[0*NPTS + tid] = (_Float16)color[oi*3+0];
    attrsT[1*NPTS + tid] = (_Float16)color[oi*3+1];
    attrsT[2*NPTS + tid] = (_Float16)color[oi*3+2];
    attrsT[3*NPTS + tid] = (_Float16)d;
    attrsT[4*NPTS + tid] = (_Float16)1.0f;
#pragma unroll
    for (int c = 5; c < 16; ++c) attrsT[c*NPTS + tid] = (_Float16)0.0f;
}

// ---------------------------------------------------------------------------
// Kernel 2: per-pixel front-to-back compositing; the 5-channel weighted
// accumulation runs on the matrix pipe via v_wmma_f32_16x16x32_f16.
// One wave = 16 pixels (A rows). Per 32-point K-slab:
//   lanes 0-15  compute w for points kb+0..7  and kb+16..23 (A K=0-7,16-23)
//   lanes 16-31 compute w for points kb+8..15 and kb+24..31 (A K=8-15,24-31)
// Transmittance ordering across the two halves is stitched with shfl_xor(16)
// of the 8-point (1-alpha) block products (exact, f32).
// Epilogue: D tile goes through a per-wave LDS staging buffer so the global
// stores are dense/coalesced instead of a divergent exec-mask ladder.
// ---------------------------------------------------------------------------
__global__ __launch_bounds__(256)
void rasterize_kernel(const float* __restrict__ params,
                      const _Float16* __restrict__ attrsT,
                      float* __restrict__ out)
{
    __shared__ __align__(16) float    s_params[NPTS*8];    // 16 KB
    __shared__ __align__(32) _Float16 s_attrs[16*NPTS];    // 16 KB
    __shared__ float                  s_tile[8][16*17];    // per-wave D staging, row-padded

    const int tid = threadIdx.x;

    // cooperative staging to LDS (128b per lane)
    {
        const uint4* gp = (const uint4*)params;
        uint4*       sp = (uint4*)s_params;
        for (int i = tid; i < (NPTS*8)/4; i += 256) sp[i] = gp[i];
        const uint4* ga = (const uint4*)attrsT;
        uint4*       sa = (uint4*)s_attrs;
        for (int i = tid; i < (16*NPTS*2)/16; i += 256) sa[i] = ga[i];
    }
    __syncthreads();

    const int lane     = tid & 31;
    const int halfid   = lane >> 4;      // 0: low half-wave, 1: high half-wave
    const int mrow     = lane & 15;      // A-matrix row (pixel) & B/D column (channel)
    const int waveInBlk= tid >> 5;
    const int waveId   = blockIdx.x * 8 + waveInBlk;
    const int pixBase  = waveId * 16;
    const int pix      = pixBase + mrow;

    const float ux = (float)(pix & (WW - 1)) + 0.5f;
    const float vy = (float)(pix >> 8) + 0.5f;

    v8f   acc = {};
    float T   = 1.0f;

    for (int kb = 0; kb < NPTS; kb += 32) {
        // B operand: 2x8 f16 attr values of my channel (K=8h..8h+7, 16+8h..16+8h+7) -> 2x ds_load_b128
        v16h Bv;
#pragma unroll
        for (int s = 0; s < 16; ++s)
            Bv[s] = s_attrs[mrow*NPTS + kb + (s >> 3)*16 + halfid*8 + (s & 7)];

        v16h Aw = {};
#pragma unroll
        for (int blk = 0; blk < 2; ++blk) {
            const int base = kb + blk*16 + halfid*8;
            float a8[8], m8[8];
            float P = 1.0f;
#pragma unroll
            for (int i = 0; i < 8; ++i) {
                const float4 p0 = *(const float4*)(&s_params[(base + i)*8]);
                const float2 p1 = *(const float2*)(&s_params[(base + i)*8 + 4]);
                const float dx = ux - p0.x;
                const float dy = vy - p0.y;
                const float power = -0.5f*(p0.z*dx*dx + p1.x*dy*dy) - p0.w*dx*dy;
                const float g  = __expf(power);
                const float pa = g * p1.y;
                const float al = (pa >= (1.0f/255.0f)) ? fminf(pa, 0.99f) : 0.0f;
                a8[i] = al;
                m8[i] = 1.0f - al;
                P *= m8[i];
            }
            // stitch front-to-back order across half-waves
            const float Pother = __shfl_xor(P, 16);
            float r = halfid ? (T * Pother) : T;
#pragma unroll
            for (int i = 0; i < 8; ++i) {
                Aw[blk*8 + i] = (_Float16)(a8[i] * r);   // w = alpha * T_excl
                r *= m8[i];
            }
            T *= P * Pother;   // both halves advance by the full 16-point product
        }

        // D(16x16 f32) += A(16x32 f16 weights) x B(32x16 f16 attrs)
        acc = __builtin_amdgcn_wmma_f32_16x16x32_f16(
            /*neg_a=*/false, Aw, /*neg_b=*/false, Bv,
            /*c_mod=*/(short)0, acc, /*reuse_a=*/false, /*reuse_b=*/false);
    }

    // ---- epilogue: stage D tile in LDS (per-wave: same-wave DS ordering,
    // no barrier needed), then emit dense coalesced stores. ----
    // D layout: VGPR r -> pixel row (halfid?8:0)+r, channel = mrow.
    {
        const int prow = halfid ? 8 : 0;
#pragma unroll
        for (int r8 = 0; r8 < 8; ++r8)
            s_tile[waveInBlk][(prow + r8)*17 + mrow] = acc[r8];
    }

    float* image     = out;                 // [HH*WW][3]
    float* depth_img = out + HH*WW*3;       // [HH*WW]
    float* acc_img   = out + HH*WW*4;       // [HH*WW]

    // image: 48 contiguous floats at image[pixBase*3], element e=(p,c) row-major
    {
        float* dst = image + pixBase*3;
        int e = lane;                                  // e in [0,32)
        dst[e] = s_tile[waveInBlk][(e/3)*17 + (e%3)];
        e = lane + 32;                                 // e in [32,48) for lanes 0-15
        if (lane < 16)
            dst[e] = s_tile[waveInBlk][(e/3)*17 + (e%3)];
    }
    // depth: lanes 0-15; acc_alpha: lanes 16-31
    if (halfid == 0)
        depth_img[pixBase + mrow] = s_tile[waveInBlk][mrow*17 + 3];
    else
        acc_img[pixBase + mrow]   = s_tile[waveInBlk][mrow*17 + 4];
}

// ---------------------------------------------------------------------------
extern "C" void kernel_launch(void* const* d_in, const int* in_sizes, int n_in,
                              void* d_out, int out_size, void* d_ws, size_t ws_size,
                              hipStream_t stream)
{
    (void)in_sizes; (void)n_in; (void)out_size; (void)ws_size;

    const float* pc    = (const float*)d_in[0];
    const float* q     = (const float*)d_in[1];
    const float* t     = (const float*)d_in[2];
    const float* Km    = (const float*)d_in[3];
    const float* color = (const float*)d_in[4];
    const float* alpha = (const float*)d_in[5];
    const float* conic = (const float*)d_in[6];

    float*    params = (float*)d_ws;                                    // 16 KB
    _Float16* attrsT = (_Float16*)((char*)d_ws + NPTS*8*sizeof(float)); // 16 KB

    prep_sort_kernel<<<1, NPTS, 0, stream>>>(pc, q, t, Km, color, alpha, conic,
                                             params, attrsT);

    // 65536 pixels / (16 pixels per wave * 8 waves per block) = 512 blocks
    rasterize_kernel<<<(HH*WW)/(16*8), 256, 0, stream>>>(params, attrsT,
                                                         (float*)d_out);
}
